// SSIMPreservationLoss_77094662963953
// MI455X (gfx1250) — matrix operand order repaired and checked by reading, not compiled, not run on hardware
//
#include <hip/hip_runtime.h>
#include <hip/hip_bf16.h>

typedef __attribute__((ext_vector_type(16))) _Float16 v16h;
typedef __attribute__((ext_vector_type(8)))  _Float16 v8h;
typedef __attribute__((ext_vector_type(8)))  float    v8f;
typedef __attribute__((ext_vector_type(4)))  unsigned u32x4;
typedef __attribute__((ext_vector_type(8)))  int      i32x8;
typedef __attribute__((ext_vector_type(4)))  int      i32x4;

#define IMG_H 512
#define IMG_W 512
#define TW 16          // tile width  (one WMMA N-group)
#define TH 64          // tile height (4 output row-groups)
#define PR 80          // patch rows  (TH + 16 halo, 5 groups of 16)
#define PC 32          // patch cols  (TW + halo, rounded to K=32)
#define LSTRIDE 34     // LDS row stride in floats (128B row + 8B pad -> no bank conflicts)

union H8U {            // packed f16 horizontal result (D layout, rows i+8*hl)
    v8h      h;
    unsigned u[4];
};
union V16U {           // full f16 A/B fragment
    v16h     h;
    unsigned u[8];
};

// Gaussian weight for tap d in [0,10] (center 5), sigma = 1.5
__device__ __forceinline__ float gauss_w(int d) {
    if (d < 0 || d > 10) return 0.0f;
    float t = (float)(d - 5);
    return __expf(t * t * (-1.0f / 4.5f)) * (1.0f / 3.7592315f);
}

// Issue one TDM 2D tile load: 80 rows x 32 f32 cols from a 512-stride plane
// into LDS at ldsByteOff, padding 2 dwords every 128B (row stride = 34 dwords).
// D# layout per CDNA5 ISA ch.8 (group0: count/lds_addr/global_addr/type,
// group1: data_size/pad/tensor dims/tile dims/strides).
__device__ __forceinline__ void tdm_load_patch(const float* gsrc, unsigned ldsByteOff)
{
    const unsigned long long ga = (unsigned long long)(uintptr_t)gsrc;
    u32x4 g0;
    g0[0] = 1u;                                               // count=1, user mode
    g0[1] = ldsByteOff;                                       // lds_addr (bytes)
    g0[2] = (unsigned)(ga & 0xffffffffull);                   // global_addr[31:0]
    g0[3] = (unsigned)((ga >> 32) & 0x1ffffffull) | (2u << 30); // addr[56:32] | type=2

    i32x8 g1;
    g1[0] = (int)((2u << 16)      // data_size = 2 -> 4 bytes
                | (1u << 20)      // pad_enable
                | (4u << 22)      // pad_interval: every (1<<4)*8B = 128B
                | (1u << 25));    // pad_amount: 2 dwords
    g1[1] = (int)((512u & 0xffffu) << 16);                    // tensor_dim0 lo16 @bit48
    g1[2] = (int)(((512u >> 16) & 0xffffu)                    // tensor_dim0 hi16
                | ((512u & 0xffffu) << 16));                  // tensor_dim1 lo16
    g1[3] = (int)(((512u >> 16) & 0xffffu)                    // tensor_dim1 hi16
                | ((unsigned)PC << 16));                      // tile_dim0 = 32
    g1[4] = (int)((unsigned)PR);                              // tile_dim1 = 80, tile_dim2=0
    g1[5] = (int)(512u);                                      // tensor_dim0_stride lo32
    g1[6] = (int)(((262144u & 0xffffu) << 16));               // stride0 hi16 | stride1 lo16
    g1[7] = (int)(262144u >> 16);                             // tensor_dim1_stride[47:16]

    const i32x4 z4 = {0, 0, 0, 0};                            // groups 2/3 unused (2D tile)
    const i32x8 z8 = {0, 0, 0, 0, 0, 0, 0, 0};
    __builtin_amdgcn_tensor_load_to_lds(g0, g1, z4, z4, z8, 0);
}

__global__ __launch_bounds__(32)
void ssim_tile_kernel(const float* __restrict__ clean,
                      const float* __restrict__ adv,
                      float* __restrict__ partial)
{
    __shared__ float ldsC[PR * LSTRIDE];
    __shared__ float ldsA[PR * LSTRIDE];

    const int lane = threadIdx.x;        // 0..31 (wave32)
    const int hl   = lane >> 4;          // register-file half (0/1)
    const int ln   = lane & 15;          // matrix row/col index within half
    const int x0   = blockIdx.x * TW;
    const int y0   = blockIdx.y * TH;
    const long planeBase = (long)blockIdx.z * IMG_H * IMG_W;

    // ---------------- stage 80x32 f32 halo patch (both images) ---------------
    const bool interior = (x0 >= 5) && (x0 + PC - 5 <= IMG_W) &&
                          (y0 >= 5) && (y0 + PR - 5 <= IMG_H);
    if (interior) {
        // Tensor Data Mover path: 2 async tensor ops stage both planes.
        const long off = planeBase + (long)(y0 - 5) * IMG_W + (x0 - 5);
        tdm_load_patch(clean + off, (unsigned)(uintptr_t)(void*)ldsC);
        tdm_load_patch(adv   + off, (unsigned)(uintptr_t)(void*)ldsA);
        __builtin_amdgcn_s_wait_tensorcnt((short)0);
    } else {
        // Border tiles: guarded scalar staging with zero padding.
        const int gx  = x0 - 5 + lane;
        const bool xok = (gx >= 0) && (gx < IMG_W);
        #pragma unroll 4
        for (int r = 0; r < PR; ++r) {
            const int gy = y0 - 5 + r;
            float c = 0.0f, a = 0.0f;
            if (xok && gy >= 0 && gy < IMG_H) {
                const long idx = planeBase + (long)gy * IMG_W + gx;
                c = clean[idx];
                a = adv[idx];
            }
            ldsC[r * LSTRIDE + lane] = c;
            ldsA[r * LSTRIDE + lane] = a;
        }
    }
    __syncthreads();

    // ------------- constant banded Gaussian fragment (B layout) --------------
    // B[j,n] = g[j-n]; elem e, lane -> K=(e<8 ? 8*hl+e : 16+8*hl+e-8), N=ln. Used by BOTH passes:
    // horizontal (h = In x G) and transposed vertical (out^T = h_win^T x G).
    v16h Bh;
    #pragma unroll
    for (int e = 0; e < 16; ++e) {
        Bh[e] = (_Float16)gauss_w(((e < 8) ? 8 * hl + e : 8 + 8 * hl + e) - ln);
    }

    // ---------------- horizontal pass: 5 row-groups x 4 signals --------------
    // Signals: c, a, (c+a)^2, (c-a)^2  (Ecc+Eaa and Eca derive from the last two)
    H8U hPk[4][5];
    #pragma unroll
    for (int g = 0; g < 5; ++g) {
        const float* rowC = &ldsC[(16 * g + ln) * LSTRIDE];
        const float* rowA = &ldsA[(16 * g + ln) * LSTRIDE];
        v16h Ac, Aa, Ap, Am;   // A fragments (16x32): elem e -> col K
        #pragma unroll
        for (int e = 0; e < 16; ++e) {
            const int v = e >> 1, slot = e & 1;
            const int K = ((v < 4) ? 0 : 16) + 8 * hl + 2 * (v & 3) + slot;
            const float c  = rowC[K];
            const float a  = rowA[K];
            const float sp = c + a;
            const float sm = c - a;
            Ac[e] = (_Float16)c;
            Aa[e] = (_Float16)a;
            Ap[e] = (_Float16)(sp * sp);
            Am[e] = (_Float16)(sm * sm);
        }
        v8f z = {};
        v8f hd[4];
        hd[0] = __builtin_amdgcn_wmma_f32_16x16x32_f16(false, Ac, false, Bh, (short)0, z, false, false);
        hd[1] = __builtin_amdgcn_wmma_f32_16x16x32_f16(false, Aa, false, Bh, (short)0, z, false, false);
        hd[2] = __builtin_amdgcn_wmma_f32_16x16x32_f16(false, Ap, false, Bh, (short)0, z, false, false);
        hd[3] = __builtin_amdgcn_wmma_f32_16x16x32_f16(false, Am, false, Bh, (short)0, z, false, false);
        #pragma unroll
        for (int s = 0; s < 4; ++s) {
            H8U p;
            #pragma unroll
            for (int j = 0; j < 8; ++j) p.h[j] = (_Float16)hd[s][j];
            hPk[s][g] = p;
        }
    }

    // --------- transposed vertical pass + SSIM formula (no shuffles) ---------
    // out^T = H_win^T x G. A-fragment of H_win^T elem e: K = base+8*hl+2v+slot
    // == window row held in this lane's own hPk registers:
    //   e<8  -> hPk[r]   element e      (rows 16r + 8hl + e)
    //   e>=8 -> hPk[r+1] element e-8    (rows 16r + 16 + 8hl + (e-8))
    const float C1 = 1e-4f, C2 = 9e-4f;
    float acc = 0.0f;
    #pragma unroll
    for (int r = 0; r < 4; ++r) {
        v8f o[4];
        #pragma unroll
        for (int s = 0; s < 4; ++s) {
            V16U Ah;
            #pragma unroll
            for (int w = 0; w < 4; ++w) {
                Ah.u[w]     = hPk[s][r].u[w];
                Ah.u[w + 4] = hPk[s][r + 1].u[w];
            }
            v8f z = {};
            o[s] = __builtin_amdgcn_wmma_f32_16x16x32_f16(false, Ah.h, false, Bh, (short)0, z, false, false);
        }
        #pragma unroll
        for (int i = 0; i < 8; ++i) {
            const float mu1 = o[0][i], mu2 = o[1][i];
            const float P   = o[2][i], M   = o[3][i];
            const float mu1s = mu1 * mu1, mu2s = mu2 * mu2, mu12 = mu1 * mu2;
            const float s12  = 0.25f * (P - M) - mu12;          // sigma12
            const float ssum = 0.5f  * (P + M) - mu1s - mu2s;   // sigma1^2 + sigma2^2
            const float num = (2.0f * mu12 + C1) * (2.0f * s12 + C2);
            const float den = (mu1s + mu2s + C1) * (ssum + C2);
            acc += __fdividef(num, den);
        }
    }

    // ---------------- wave reduction, store per-WG partial -------------------
    #pragma unroll
    for (int off = 16; off > 0; off >>= 1)
        acc += __shfl_xor(acc, off, 32);
    if (lane == 0) {
        const int wg = (blockIdx.z * gridDim.y + blockIdx.y) * gridDim.x + blockIdx.x;
        partial[wg] = acc;
    }
}

__global__ __launch_bounds__(256)
void ssim_finalize(const float* __restrict__ partial, int n, float* __restrict__ out)
{
    __shared__ double sdata[256];
    double s = 0.0;
    for (int i = threadIdx.x; i < n; i += 256) s += (double)partial[i];
    sdata[threadIdx.x] = s;
    __syncthreads();
    for (int stride = 128; stride > 0; stride >>= 1) {
        if (threadIdx.x < stride) sdata[threadIdx.x] += sdata[threadIdx.x + stride];
        __syncthreads();
    }
    if (threadIdx.x == 0) {
        const double total = 32.0 * 3.0 * 512.0 * 512.0;
        out[0] = (float)(1.0 - sdata[0] / total);
    }
}

extern "C" void kernel_launch(void* const* d_in, const int* in_sizes, int n_in,
                              void* d_out, int out_size, void* d_ws, size_t ws_size,
                              hipStream_t stream) {
    const float* clean = (const float*)d_in[0];
    const float* adv   = (const float*)d_in[1];
    float* partial     = (float*)d_ws;

    dim3 grid(IMG_W / TW, IMG_H / TH, 32 * 3);   // 32 x 8 x 96 tiles, 1 wave each
    ssim_tile_kernel<<<grid, 32, 0, stream>>>(clean, adv, partial);

    const int nPartial = (IMG_W / TW) * (IMG_H / TH) * 32 * 3;
    ssim_finalize<<<1, 256, 0, stream>>>(partial, nPartial, (float*)d_out);
}
